// GraphAdapter_67628555042877
// MI455X (gfx1250) — matrix operand order repaired and checked by reading, not compiled, not run on hardware
//
#include <hip/hip_runtime.h>
#include <hip/hip_bf16.h>
#include <stdint.h>

// ---------------- types ----------------
typedef __bf16 bhalf;
typedef __attribute__((ext_vector_type(16))) __bf16 v16bf;
typedef __attribute__((ext_vector_type(8)))  __bf16 v8bf;
typedef __attribute__((ext_vector_type(8)))  float  v8f;

#define HIDDEN 64
#define HEADS  8
#define OUTC   8
#define LLMD   4096
#define EPSR   1e-6f
#define SCALE_ATT 0.35355339059327373f  // 1/sqrt(8)

__device__ __forceinline__ float silu_f(float x) {
    return x * (1.0f / (1.0f + __expf(-x)));
}

// sign-split float atomic max (works for mixed signs, init = -inf)
__device__ __forceinline__ void atomicMaxF(float* a, float v) {
    if (v >= 0.0f) atomicMax((int*)a, __float_as_int(v));
    else           atomicMin((unsigned int*)a, __float_as_uint(v));
}

// scheduler fence: keep LDS fragment loads batched ahead of the WMMA chain
__device__ __forceinline__ void sched_fence() {
#if __has_builtin(__builtin_amdgcn_sched_barrier)
    __builtin_amdgcn_sched_barrier(0);
#else
    asm volatile("" ::: "memory");
#endif
}

// ---------------- utility kernels ----------------
// Convert fp32 weights [K,N] to bf16 in K-panel-major layout:
// out[(k/32)*N*32 + n*32 + (k%32)]  -- each 32-row K panel is a contiguous
// col-major [N][32] block, so GEMM B staging is a straight async memcpy.
__global__ __launch_bounds__(256) void k_cvt_bf16_panel(const float* __restrict__ in,
                                                        bhalf* __restrict__ out,
                                                        int K, int N) {
    int i = blockIdx.x * 256 + threadIdx.x;
    if (i >= K * N) return;
    int k = i / N, n = i % N;
    out[(size_t)(k >> 5) * N * 32 + n * 32 + (k & 31)] = (bhalf)in[i];
}

__global__ __launch_bounds__(256) void k_fill(float* __restrict__ p, float v, int n) {
    int i = blockIdx.x * 256 + threadIdx.x;
    if (i < n) p[i] = v;
}

// ---------------- WMMA bf16 GEMM: C[M,N] = act(A_fp32[M,K] @ Bpanel_bf16[K,N]) ----------------
// block = 256 threads (8 waves), block tile = 128 x BN, BN == N (64 or 128), K % 32 == 0.
// B must be in K-panel-major bf16 layout (see k_cvt_bf16_panel).
// Double-buffered LDS: one barrier per k-step; async B copy + A loads for k+1
// overlap the BN/16-deep (4 or 8) back-to-back WMMA chain of step k.
template<int BN, bool DO_SILU>
__global__ __launch_bounds__(256) void k_gemm_wmma(
        const float* __restrict__ A, const bhalf* __restrict__ B,
        float* __restrict__ C, int M, int K, int ldc, int coloff)
{
    __shared__ alignas(16) bhalf As[2 * 128 * 32];   // ping-pong row-major [128][32]
    __shared__ alignas(16) bhalf Bs[2 * BN * 32];    // ping-pong col-major [BN][32]

    const int tid  = threadIdx.x;
    const int wave = tid >> 5;
    const int lane = tid & 31;
    const int row  = lane & 15;      // M row within 16-tile (N col for B/C)
    const int half = lane >> 4;      // hi/lo lane group
    const int m0   = blockIdx.x * 128;

    v8f acc[BN / 16];
#pragma unroll
    for (int t = 0; t < BN / 16; t++) acc[t] = (v8f){0,0,0,0,0,0,0,0};

    const int ar = tid >> 1;                 // A stage: row, 2 threads per row
    const int ac = (tid & 1) * 16;           // 16 contiguous floats each
    const bool arow_ok = (m0 + ar) < M;
    const float* aptr = A + (size_t)(m0 + ar) * K + ac;

    // issue async copy of one contiguous 32xBN bf16 panel into LDS buffer `buf`
    auto issueB = [&](int k0, int buf) {
        const bhalf* bpanel = B + (size_t)(k0 >> 5) * (BN * 32);
#pragma unroll
        for (int j = 0; j < BN / 64; j++) {      // 16B chunks per thread (2 or 1)
            int c = tid + j * 256;
            unsigned lp = (unsigned)(uintptr_t)(const void*)&Bs[buf * (BN * 32) + c * 8];
            unsigned long long gp = (unsigned long long)(uintptr_t)(bpanel + c * 8);
            asm volatile("global_load_async_to_lds_b128 %0, %1, off"
                         :: "v"(lp), "v"(gp) : "memory");
        }
    };
    // fp32 -> bf16 convert + packed dual ds_store_b128 into LDS buffer `buf`
    auto stageA = [&](int k0, int buf) {
        union PK { v8bf v; bhalf e[8]; } p0, p1;
        if (arow_ok) {
            const float4* ap = (const float4*)(aptr + k0);
            float4 v0 = ap[0], v1 = ap[1], v2 = ap[2], v3 = ap[3];
            p0.e[0]=(bhalf)v0.x; p0.e[1]=(bhalf)v0.y; p0.e[2]=(bhalf)v0.z; p0.e[3]=(bhalf)v0.w;
            p0.e[4]=(bhalf)v1.x; p0.e[5]=(bhalf)v1.y; p0.e[6]=(bhalf)v1.z; p0.e[7]=(bhalf)v1.w;
            p1.e[0]=(bhalf)v2.x; p1.e[1]=(bhalf)v2.y; p1.e[2]=(bhalf)v2.z; p1.e[3]=(bhalf)v2.w;
            p1.e[4]=(bhalf)v3.x; p1.e[5]=(bhalf)v3.y; p1.e[6]=(bhalf)v3.z; p1.e[7]=(bhalf)v3.w;
        } else {
#pragma unroll
            for (int i = 0; i < 8; i++) { p0.e[i] = (bhalf)0.0f; p1.e[i] = (bhalf)0.0f; }
        }
        v8bf* dst = (v8bf*)&As[buf * (128 * 32) + ar * 32 + ac];
        dst[0] = p0.v;
        dst[1] = p1.v;
    };

    union UF { v16bf v; v8bf h[2]; };

    // prologue: stage k=0 into buffer 0
    issueB(0, 0);
    stageA(0, 0);

    for (int k0 = 0; k0 < K; k0 += 32) {
        const int cur = (k0 >> 5) & 1;
        // own async chunks for `cur` complete; implicit s_wait_dscnt before the
        // barrier covers own ds stores/loads; barrier publishes `cur` WGP-wide.
        asm volatile("s_wait_asynccnt 0x0" ::: "memory");
        __syncthreads();

        // overlap: kick off k+1 staging into the other buffer
        if (k0 + 32 < K) {
            issueB(k0 + 32, cur ^ 1);
            stageA(k0 + 32, cur ^ 1);
        }

        const bhalf* Asb = &As[cur * (128 * 32)];
        const bhalf* Bsb = &Bs[cur * (BN * 32)];

        // A fragment: lane holds row (m0+wave*16+row), K = [16h..16h+15]
        UF ua;
        {
            const bhalf* abase = Asb + (wave * 16 + row) * 32;
            ua.h[0] = *(const v8bf*)(abase + 16 * half);
            ua.h[1] = *(const v8bf*)(abase + 16 * half + 8);
        }
        // preload ALL B fragments...
        UF ub[BN / 16];
#pragma unroll
        for (int nt = 0; nt < BN / 16; nt++) {
            const bhalf* bbase = Bsb + (nt * 16 + row) * 32 + 16 * half;
            ub[nt].h[0] = *(const v8bf*)(bbase);
            ub[nt].h[1] = *(const v8bf*)(bbase + 8);
        }
        // ...fence so the loads stay batched (one DS wait), then issue the
        // WMMAs back-to-back (ISA 7.5 XDL pipelining)
        sched_fence();
#pragma unroll
        for (int nt = 0; nt < BN / 16; nt++) {
            acc[nt] = __builtin_amdgcn_wmma_f32_16x16x32_bf16(
                false, ua.v, false, ub[nt].v, (short)0, acc[nt], false, false);
        }
        // no trailing barrier: next iteration's barrier protects both buffers
    }

    // ---- store C: VGPR i -> row = i + 8*half, col = lane&15 (ISA 16x16 f32 C/D layout)
#pragma unroll
    for (int nt = 0; nt < BN / 16; nt++) {
#pragma unroll
        for (int i = 0; i < 8; i++) {
            int r = m0 + wave * 16 + i + 8 * half;
            if (r < M) {
                float v = acc[nt][i];
                if (DO_SILU) v = silu_f(v);
                C[(size_t)r * ldc + coloff + nt * 16 + row] = v;
            }
        }
    }
}

// ---------------- small fused GEMM (N = 64): out = [norm](act(in[nrows,K] @ W[K,64] (+bias))) ----------------
// block = 256 threads = 4 rows x 64 cols
template<int K, bool DO_SILU, bool DO_NORM, bool DO_BIAS, bool DO_GATHER>
__global__ __launch_bounds__(256) void k_gemm_small(
        const float* __restrict__ In, const float* __restrict__ W,
        const float* __restrict__ bias, const float* __restrict__ wn,
        const int* __restrict__ gather, float* __restrict__ Out,
        int nrows, int ldc, int coloff)
{
    __shared__ float Ws[K * 64];
    __shared__ float Rs[4 * K];
    __shared__ float red[256];

    const int tid = threadIdx.x;
    const int lr  = tid >> 6;
    const int col = tid & 63;

    for (int i = tid; i < K * 64; i += 256) Ws[i] = W[i];

    const int r = blockIdx.x * 4 + lr;
    int rr = 0;
    if (r < nrows) rr = DO_GATHER ? gather[r] : r;
    for (int i = col; i < K; i += 64) Rs[lr * K + i] = In[(size_t)rr * K + i];
    __syncthreads();

    float acc = DO_BIAS ? bias[col] : 0.0f;
#pragma unroll 8
    for (int k = 0; k < K; k++) acc += Rs[lr * K + k] * Ws[k * 64 + col];
    if (DO_SILU) acc = silu_f(acc);

    if (DO_NORM) {
        red[tid] = acc * acc;
        __syncthreads();
#pragma unroll
        for (int s = 32; s > 0; s >>= 1) {
            if (col < s) red[tid] += red[tid + s];
            __syncthreads();
        }
        float var = red[lr * 64] * (1.0f / 64.0f);
        acc = wn[col] * acc * rsqrtf(var + EPSR);
    }
    if (r < nrows) Out[(size_t)r * ldc + coloff + col] = acc;
}

// ---------------- GAT edge kernels (one thread per (edge, head)) ----------------
__global__ __launch_bounds__(256) void k_edge_alpha_max(
        const float* __restrict__ XL, const float* __restrict__ XR,
        const int* __restrict__ src, const int* __restrict__ dst,
        float* __restrict__ alpha, float* __restrict__ Mx, int nEh)
{
    int i = blockIdx.x * 256 + threadIdx.x;
    if (i >= nEh) return;
    int e = i >> 3, h = i & 7;
    int s = src[e], d = dst[e];
    const float* xi = XL + (size_t)d * HIDDEN + h * OUTC;
    const float* xj = XR + (size_t)s * HIDDEN + h * OUTC;
    float a = 0.0f;
#pragma unroll
    for (int c = 0; c < OUTC; c++) a += xi[c] * xj[c];
    a *= SCALE_ATT;
    alpha[i] = a;
    atomicMaxF(&Mx[(size_t)d * HEADS + h], a);
}

__global__ __launch_bounds__(256) void k_edge_exp_sum(
        const int* __restrict__ dst, float* __restrict__ alpha,
        const float* __restrict__ Mx, float* __restrict__ Sx, int nEh)
{
    int i = blockIdx.x * 256 + threadIdx.x;
    if (i >= nEh) return;
    int e = i >> 3, h = i & 7;
    int d = dst[e];
    float ex = __expf(alpha[i] - Mx[(size_t)d * HEADS + h]);
    alpha[i] = ex;
    atomicAdd(&Sx[(size_t)d * HEADS + h], ex);
}

__global__ __launch_bounds__(256) void k_edge_scatter(
        const int* __restrict__ src, const int* __restrict__ dst,
        const float* __restrict__ alpha, const float* __restrict__ Sx,
        const float* __restrict__ XR, float* __restrict__ AGG, int nEh)
{
    int i = blockIdx.x * 256 + threadIdx.x;
    if (i >= nEh) return;
    int e = i >> 3, h = i & 7;
    int s = src[e], d = dst[e];
    float w = alpha[i] / (Sx[(size_t)d * HEADS + h] + 1e-16f);
    const float* xj = XR + (size_t)s * HIDDEN + h * OUTC;
    float* ag = AGG + (size_t)d * HIDDEN + h * OUTC;
#pragma unroll
    for (int c = 0; c < OUTC; c++) atomicAdd(&ag[c], xj[c] * w);
}

// ---------------- generic VALU GEMM (wide N), thread per output element ----------------
template<bool DO_SILU>
__global__ __launch_bounds__(256) void k_gemm_valu(
        const float* __restrict__ In, const float* __restrict__ W,
        float* __restrict__ Out, int rows, int Kd, int Nd)
{
    int i = blockIdx.x * 256 + threadIdx.x;
    if (i >= rows * Nd) return;
    int b = i / Nd, n = i % Nd;
    const float* ir = In + (size_t)b * Kd;
    float acc = 0.0f;
#pragma unroll 8
    for (int k = 0; k < Kd; k++) acc += ir[k] * W[(size_t)k * Nd + n];
    Out[i] = DO_SILU ? silu_f(acc) : acc;
}

// ---------------- host-side pipeline ----------------
static inline int ceil_div(long long a, long long b) { return (int)((a + b - 1) / b); }

extern "C" void kernel_launch(void* const* d_in, const int* in_sizes, int n_in,
                              void* d_out, int out_size, void* d_ws, size_t ws_size,
                              hipStream_t stream)
{
    const float* x        = (const float*)d_in[0];
    const int*   edge     = (const int*)  d_in[1];
    const int*   node_ids = (const int*)  d_in[2];
    const float* prompt   = (const float*)d_in[3];
    const float* W_lin1   = (const float*)d_in[4];
    const float* W_lin2   = (const float*)d_in[5];
    const float* w_bn1    = (const float*)d_in[6];
    const float* w_bn2    = (const float*)d_in[7];
    const float* w_bn3    = (const float*)d_in[8];
    const float* W_gnn1   = (const float*)d_in[9];
    const float* W_gnn2   = (const float*)d_in[10];
    const float* W_attl   = (const float*)d_in[11];
    const float* b_attl   = (const float*)d_in[12];
    const float* W_attr   = (const float*)d_in[13];
    const float* b_attr   = (const float*)d_in[14];
    const float* W_attl1  = (const float*)d_in[15];
    const float* b_attl1  = (const float*)d_in[16];
    const float* W_attr1  = (const float*)d_in[17];
    const float* b_attr1  = (const float*)d_in[18];
    const float* W_prompt = (const float*)d_in[19];
    const float* W_g      = (const float*)d_in[20];
    const float* W_f1     = (const float*)d_in[21];
    const float* W_f2     = (const float*)d_in[22];
    const float* W_ext    = (const float*)d_in[23];

    const int N = in_sizes[0] / LLMD;        // 50000 nodes
    const int E = in_sizes[1] / 2;           // 1.6M edges
    const int Bq = in_sizes[2];              // 4096 queries
    const int* src = edge;
    const int* dst = edge + E;

    // workspace carving (256B aligned)
    uintptr_t cur = (uintptr_t)d_ws;
    auto carve = [&](size_t bytes) -> void* {
        void* p = (void*)cur;
        cur += (bytes + 255) & ~(size_t)255;
        return p;
    };
    bhalf* W1bf  = (bhalf*)carve((size_t)LLMD * 128 * sizeof(bhalf));
    bhalf* Wpbf  = (bhalf*)carve((size_t)LLMD * 64 * sizeof(bhalf));
    float* H1    = (float*)carve((size_t)N * 128 * sizeof(float));
    float* Hc    = (float*)carve((size_t)N * 64 * sizeof(float));
    float* XL    = (float*)carve((size_t)N * 64 * sizeof(float));
    float* XR    = (float*)carve((size_t)N * 64 * sizeof(float));
    float* ALPHA = (float*)carve((size_t)E * HEADS * sizeof(float));
    float* Mx    = (float*)carve((size_t)N * HEADS * sizeof(float));
    float* Sx    = (float*)carve((size_t)N * HEADS * sizeof(float));
    float* AGG   = (float*)carve((size_t)N * 64 * sizeof(float));
    float* F0    = (float*)carve((size_t)Bq * 128 * sizeof(float));
    float* F1b   = (float*)carve((size_t)Bq * 640 * sizeof(float));
    float* F2b   = (float*)carve((size_t)Bq * 64 * sizeof(float));
    (void)ws_size;

    const dim3 blk(256);
    const int nEh = E * HEADS;

    // weight conversion to bf16, K-panel-major layout for async LDS staging
    k_cvt_bf16_panel<<<ceil_div((long long)LLMD * 128, 256), blk, 0, stream>>>(W_lin1, W1bf, LLMD, 128);
    k_cvt_bf16_panel<<<ceil_div((long long)LLMD * 64, 256), blk, 0, stream>>>(W_prompt, Wpbf, LLMD, 64);

    // 1) H1 = silu(x @ W_lin1)    [N,128]  -- dominant GEMM, bf16 WMMA + async-to-LDS
    k_gemm_wmma<128, true><<<ceil_div(N, 128), blk, 0, stream>>>(x, W1bf, H1, N, LLMD, 128, 0);

    // 2) Hc = rmsnorm(H1 @ W_lin2, w_bn1)  [N,64]
    k_gemm_small<128, false, true, false, false><<<ceil_div(N, 4), blk, 0, stream>>>(
        H1, W_lin2, nullptr, w_bn1, nullptr, Hc, N, 64, 0);

    // ---- GAT propagate #1 ----
    k_gemm_small<64, false, false, true, false><<<ceil_div(N, 4), blk, 0, stream>>>(
        Hc, W_attl, b_attl, nullptr, nullptr, XL, N, 64, 0);
    k_gemm_small<64, false, false, true, false><<<ceil_div(N, 4), blk, 0, stream>>>(
        Hc, W_attr, b_attr, nullptr, nullptr, XR, N, 64, 0);
    k_fill<<<ceil_div((long long)N * HEADS, 256), blk, 0, stream>>>(Mx, -INFINITY, N * HEADS);
    k_fill<<<ceil_div((long long)N * HEADS, 256), blk, 0, stream>>>(Sx, 0.0f, N * HEADS);
    k_fill<<<ceil_div((long long)N * 64, 256), blk, 0, stream>>>(AGG, 0.0f, N * 64);
    k_edge_alpha_max<<<ceil_div(nEh, 256), blk, 0, stream>>>(XL, XR, src, dst, ALPHA, Mx, nEh);
    k_edge_exp_sum<<<ceil_div(nEh, 256), blk, 0, stream>>>(dst, ALPHA, Mx, Sx, nEh);
    k_edge_scatter<<<ceil_div(nEh, 256), blk, 0, stream>>>(src, dst, ALPHA, Sx, XR, AGG, nEh);

    // 3) Hc = rmsnorm(silu(AGG @ W_gnn1), w_bn2)
    k_gemm_small<64, true, true, false, false><<<ceil_div(N, 4), blk, 0, stream>>>(
        AGG, W_gnn1, nullptr, w_bn2, nullptr, Hc, N, 64, 0);

    // ---- GAT propagate #2 ----
    k_gemm_small<64, false, false, true, false><<<ceil_div(N, 4), blk, 0, stream>>>(
        Hc, W_attl1, b_attl1, nullptr, nullptr, XL, N, 64, 0);
    k_gemm_small<64, false, false, true, false><<<ceil_div(N, 4), blk, 0, stream>>>(
        Hc, W_attr1, b_attr1, nullptr, nullptr, XR, N, 64, 0);
    k_fill<<<ceil_div((long long)N * HEADS, 256), blk, 0, stream>>>(Mx, -INFINITY, N * HEADS);
    k_fill<<<ceil_div((long long)N * HEADS, 256), blk, 0, stream>>>(Sx, 0.0f, N * HEADS);
    k_fill<<<ceil_div((long long)N * 64, 256), blk, 0, stream>>>(AGG, 0.0f, N * 64);
    k_edge_alpha_max<<<ceil_div(nEh, 256), blk, 0, stream>>>(XL, XR, src, dst, ALPHA, Mx, nEh);
    k_edge_exp_sum<<<ceil_div(nEh, 256), blk, 0, stream>>>(dst, ALPHA, Mx, Sx, nEh);
    k_edge_scatter<<<ceil_div(nEh, 256), blk, 0, stream>>>(src, dst, ALPHA, Sx, XR, AGG, nEh);

    // 4) gx = rmsnorm(AGG @ W_gnn2, w_bn3)  -> Hc
    k_gemm_small<64, false, true, false, false><<<ceil_div(N, 4), blk, 0, stream>>>(
        AGG, W_gnn2, nullptr, w_bn3, nullptr, Hc, N, 64, 0);

    // ---- FusionBlock ----
    // token = prompt @ W_prompt -> F0[:, 64:128]   (bf16 WMMA, K=4096)
    k_gemm_wmma<64, false><<<ceil_div(Bq, 128), blk, 0, stream>>>(prompt, Wpbf, F0, Bq, LLMD, 128, 64);
    // out0 = gx[node_ids] @ W_g -> F0[:, 0:64]
    k_gemm_small<64, false, false, false, true><<<ceil_div(Bq, 4), blk, 0, stream>>>(
        Hc, W_g, nullptr, nullptr, node_ids, F0, Bq, 128, 0);
    // F1 = silu(F0 @ W_f1)   [Bq,640]
    k_gemm_valu<true><<<ceil_div((long long)Bq * 640, 256), blk, 0, stream>>>(F0, W_f1, F1b, Bq, 128, 640);
    // F2 = F1 @ W_f2         [Bq,64]
    k_gemm_valu<false><<<ceil_div((long long)Bq * 64, 256), blk, 0, stream>>>(F1b, W_f2, F2b, Bq, 640, 64);
    // out = F2 @ W_ext       [Bq,4096] -> d_out
    k_gemm_valu<false><<<ceil_div((long long)Bq * LLMD, 256), blk, 0, stream>>>(
        F2b, W_ext, (float*)d_out, Bq, 64, LLMD);
}
